// Block_26371099197460
// MI455X (gfx1250) — matrix operand order repaired and checked by reading, not compiled, not run on hardware
//
#include <hip/hip_runtime.h>
#include <hip/hip_bf16.h>
#include <math.h>
#include <stdint.h>

// ---------------------------------------------------------------------------
// GPT-2 block on gfx1250 (MI455X), bf16 dataflow:
//   - weights transposed+converted to bf16 [N][K] once per launch
//   - LN / GEMM / attention produce bf16 activations directly
//   - double-buffered GLOBAL_LOAD_ASYNC_TO_LDS_B128 pipeline (ASYNCcnt)
//     overlapping DMA with v_wmma_f32_16x16x32_bf16 issue
// ---------------------------------------------------------------------------

typedef __attribute__((ext_vector_type(16))) __bf16 v16bf;
typedef __attribute__((ext_vector_type(8)))  __bf16 v8bf;
typedef __attribute__((ext_vector_type(8)))  float  v8f;

#define N_EMBD   768
#define N_HEAD   12
#define HEAD_DIM 64
#define BB       2
#define TT       2048
#define C3       (3 * N_EMBD)
#define C4       (4 * N_EMBD)
#define ROWS     (BB * TT)   // 4096

__device__ __forceinline__ __bf16 f2bf(float f) { return (__bf16)f; }

__device__ __forceinline__ v8f wmma_bf16(v16bf a, v16bf b, v8f c) {
  return __builtin_amdgcn_wmma_f32_16x16x32_bf16(false, a, false, b,
                                                 (short)0, c, false, false);
}

// Async 16B copy global -> LDS (ASYNCcnt). ldsoff = raw LDS byte offset.
__device__ __forceinline__ void async_copy16(uint32_t ldsoff, const void* g) {
  asm volatile("global_load_async_to_lds_b128 %0, %1, off"
               :: "v"(ldsoff), "v"((uint64_t)(uintptr_t)g) : "memory");
}
__device__ __forceinline__ void async_wait() {
  asm volatile("s_wait_asynccnt 0" ::: "memory");
}
__device__ __forceinline__ uint32_t lds_addr(const void* p) {
  return (uint32_t)(uintptr_t)p;
}

// A-fragment (16-bit A 16x32): lane row = lane%16, kb = (lane/16)*8,
// elems 0..7 -> K=kb..kb+7, elems 8..15 -> K=kb+16..kb+23.
__device__ __forceinline__ v16bf load_fragA(const __bf16* p) {
  v8bf lo = *reinterpret_cast<const v8bf*>(p);
  v8bf hi = *reinterpret_cast<const v8bf*>(p + 16);
  v16bf r;
#pragma unroll
  for (int i = 0; i < 8; i++) { r[i] = lo[i]; r[i + 8] = hi[i]; }
  return r;
}
// B-fragment (32x16, N-major): lane col = lane%16, kb = (lane/16)*8,
// elems 0..7 -> K=kb..kb+7, elems 8..15 -> K=kb+16..kb+23.
__device__ __forceinline__ v16bf load_fragB(const __bf16* p) {
  v8bf lo = *reinterpret_cast<const v8bf*>(p);
  v8bf hi = *reinterpret_cast<const v8bf*>(p + 16);
  v16bf r;
#pragma unroll
  for (int i = 0; i < 8; i++) { r[i] = lo[i]; r[i + 8] = hi[i]; }
  return r;
}

__device__ __forceinline__ float redmax16(float v) {
#pragma unroll
  for (int m = 1; m < 16; m <<= 1) v = fmaxf(v, __shfl_xor(v, m, 32));
  return v;
}
__device__ __forceinline__ float redsum16(float v) {
#pragma unroll
  for (int m = 1; m < 16; m <<= 1) v += __shfl_xor(v, m, 32);
  return v;
}

// ---------------------------------------------------------------------------
// Weight convert + transpose: W[K][N] f32 -> Wt[N][K] bf16 (once per launch).
// ---------------------------------------------------------------------------
__global__ __launch_bounds__(256) void wconv_kernel(const float* __restrict__ W,
                                                    __bf16* __restrict__ Wt,
                                                    int K, int N) {
  int idx = blockIdx.x * 256 + threadIdx.x;
  int n = idx / K, k = idx - n * K;
  Wt[(size_t)n * K + k] = f2bf(W[(size_t)k * N + n]);
}

// ---------------------------------------------------------------------------
// LayerNorm: f32 in, bf16 out. One 256-thread block per row of 768.
// ---------------------------------------------------------------------------
__global__ __launch_bounds__(256) void ln_kernel(const float* __restrict__ x,
                                                 const float* __restrict__ g,
                                                 const float* __restrict__ b,
                                                 __bf16* __restrict__ out) {
  int row = blockIdx.x;
  int t = threadIdx.x;
  const float* xr = x + (size_t)row * N_EMBD;
  float v0 = xr[t], v1 = xr[t + 256], v2 = xr[t + 512];

  __shared__ float sbuf[256];
  sbuf[t] = v0 + v1 + v2;
  __syncthreads();
#pragma unroll
  for (int off = 128; off > 0; off >>= 1) {
    if (t < off) sbuf[t] += sbuf[t + off];
    __syncthreads();
  }
  float mu = sbuf[0] * (1.0f / N_EMBD);
  __syncthreads();

  float d0 = v0 - mu, d1 = v1 - mu, d2 = v2 - mu;
  sbuf[t] = d0 * d0 + d1 * d1 + d2 * d2;
  __syncthreads();
#pragma unroll
  for (int off = 128; off > 0; off >>= 1) {
    if (t < off) sbuf[t] += sbuf[t + off];
    __syncthreads();
  }
  float rstd = rsqrtf(sbuf[0] * (1.0f / N_EMBD) + 1e-5f);

  __bf16* orow = out + (size_t)row * N_EMBD;
  orow[t]       = f2bf(d0 * rstd * g[t]       + b[t]);
  orow[t + 256] = f2bf(d1 * rstd * g[t + 256] + b[t + 256]);
  orow[t + 512] = f2bf(d2 * rstd * g[t + 512] + b[t + 512]);
}

// ---------------------------------------------------------------------------
// GEMM: acc = A[M,K](bf16) @ Bt[N,K](bf16)^T + bias.
// Block tile 128x64, BK=64, 512 threads = 16 waves, 32x16 output per wave
// (2 accumulators, shared B fragment). Double-buffered async-to-LDS pipeline.
// MODE 0: out f32 (+ optional residual)
// MODE 1: GELU -> out bf16
// MODE 2: qkv epilogue: q scaled by 1/8, out bf16, V also stored [b,h,d,t]
// ---------------------------------------------------------------------------
template <int MODE>
__global__ __launch_bounds__(512) void gemm_kernel(
    const __bf16* __restrict__ A, const __bf16* __restrict__ Bt,
    const float* __restrict__ bias, const float* __restrict__ res,
    float* __restrict__ outf, __bf16* __restrict__ outb,
    __bf16* __restrict__ vT, int M, int N, int K) {
  __shared__ __align__(16) __bf16 ldsA[2][128][64];   // [buf][m][k]  16KB each
  __shared__ __align__(16) __bf16 ldsB[2][64][64];    // [buf][n][k]   8KB each

  int tid  = threadIdx.x;
  int lane = tid & 31;
  int wave = tid >> 5;
  int m0 = blockIdx.y * 128;
  int n0 = blockIdx.x * 64;
  int wm = (wave >> 2) * 32;
  int wn = (wave & 3) * 16;

  int fr  = lane & 15;
  int kbA = (lane >> 4) * 8;
  int kbB = (lane >> 4) * 8;

  // Copy mapping: 16B chunks (8 bf16). A tile = 1024 chunks (2/thread),
  // B tile = 512 chunks (1/thread).
  int ca0 = tid * 2, ca1 = tid * 2 + 1;
  uint32_t aL0 = lds_addr(&ldsA[0][0][0]) + ca0 * 16;
  uint32_t aL1 = lds_addr(&ldsA[0][0][0]) + ca1 * 16;
  uint32_t bL  = lds_addr(&ldsB[0][0][0]) + tid * 16;
  const __bf16* gA0 = A + (size_t)(m0 + (ca0 >> 3)) * K + (ca0 & 7) * 8;
  const __bf16* gA1 = A + (size_t)(m0 + (ca1 >> 3)) * K + (ca1 & 7) * 8;
  const __bf16* gB  = Bt + (size_t)(n0 + (tid >> 3)) * K + (tid & 7) * 8;

  v8f acc0 = {}, acc1 = {};
  int nk = K / 64;

  // Prologue: tile 0 -> buffer 0.
  async_copy16(aL0, gA0);
  async_copy16(aL1, gA1);
  async_copy16(bL,  gB);

  for (int i = 0; i < nk; i++) {
    async_wait();            // tile i copies (this wave's) complete
    __syncthreads();         // tile i visible block-wide
    if (i + 1 < nk) {        // kick off tile i+1 while computing tile i
      int nb = (i + 1) & 1;
      async_copy16(aL0 + nb * 16384, gA0 + (i + 1) * 64);
      async_copy16(aL1 + nb * 16384, gA1 + (i + 1) * 64);
      async_copy16(bL  + nb * 8192,  gB  + (i + 1) * 64);
    }
    int cb = i & 1;
#pragma unroll
    for (int kk = 0; kk < 64; kk += 32) {
      v16bf a0 = load_fragA(&ldsA[cb][wm + fr][kk + kbA]);
      v16bf a1 = load_fragA(&ldsA[cb][wm + 16 + fr][kk + kbA]);
      v16bf b  = load_fragB(&ldsB[cb][wn + fr][kk + kbB]);
      acc0 = wmma_bf16(a0, b, acc0);
      acc1 = wmma_bf16(a1, b, acc1);
    }
    __syncthreads();         // reads done before buffer reuse at i+2
  }

  // Epilogue. C/D layout: lane col = lane%16; elem r -> row r + (lane>=16)*8.
  int n = n0 + wn + fr;
  float bn = bias[n];
#pragma unroll
  for (int t2 = 0; t2 < 2; t2++) {
    v8f acc = t2 ? acc1 : acc0;
#pragma unroll
    for (int r = 0; r < 8; r++) {
      int m = m0 + wm + t2 * 16 + r + ((lane >> 4) << 3);
      float v = acc[r] + bn;
      if (MODE == 0) {
        if (res) v += res[(size_t)m * N + n];
        outf[(size_t)m * N + n] = v;
      } else if (MODE == 1) {
        v = 0.5f * v * (1.0f + erff(v * 0.70710678118f));
        outb[(size_t)m * N + n] = f2bf(v);
      } else {  // MODE 2: qkv
        if (n < N_EMBD) v *= 0.125f;  // fold 1/sqrt(64) into Q
        outb[(size_t)m * N + n] = f2bf(v);
        if (n >= 2 * N_EMBD) {        // V -> [b, h, d, t] for attention
          int bb = m >> 11, t = m & (TT - 1);
          int hd = n - 2 * N_EMBD;
          int h = hd >> 6, d = hd & 63;
          vT[(((size_t)bb * N_HEAD + h) * HEAD_DIM + d) * TT + t] = f2bf(v);
        }
      }
    }
  }
}

// ---------------------------------------------------------------------------
// Flash attention (causal). Grid (T/64, H, B), 128 threads = 4 waves; each
// wave owns 16 query rows; 32-key tiles, double-buffered async-to-LDS.
// qkvb: bf16 [B*T, 3C] (Q pre-scaled); vT: bf16 [B,H,D,T]; y: bf16 [B*T, C].
// ---------------------------------------------------------------------------
__global__ __launch_bounds__(128) void attn_kernel(
    const __bf16* __restrict__ qkvb, const __bf16* __restrict__ vT,
    __bf16* __restrict__ y) {
  int b = blockIdx.z, h = blockIdx.y, qt = blockIdx.x;
  int tid = threadIdx.x, lane = tid & 31, wave = tid >> 5;
  int qbase = qt * 64;
  int qrow0 = qbase + wave * 16;
  int fr    = lane & 15;
  int half8 = (lane >> 4) << 3;
  int kbA   = (lane >> 4) * 8;
  int kbB   = (lane >> 4) * 8;

  __shared__ __align__(16) __bf16 ldsK[2][32][64];    // [buf][key][d]  4KB each
  __shared__ __align__(16) __bf16 ldsV[2][64][32];    // [buf][d][key]  4KB each
  __shared__ __align__(16) __bf16 ldsP[4][16][32];    // per-wave P tile

  const __bf16* qp = qkvb + (size_t)(b * TT) * C3 + h * HEAD_DIM;
  const __bf16* kp = qp + N_EMBD;
  const __bf16* vp = vT + ((size_t)b * N_HEAD + h) * HEAD_DIM * TT;

  // Q fragments straight from bf16 global (Q already scaled by 1/8).
  v16bf qf[2];
#pragma unroll
  for (int s = 0; s < 2; s++) {
    const __bf16* qr = qp + (size_t)(qrow0 + fr) * C3 + s * 32;
    v8bf lo = *reinterpret_cast<const v8bf*>(qr + kbA);
    v8bf hi = *reinterpret_cast<const v8bf*>(qr + kbA + 16);
    v16bf a;
#pragma unroll
    for (int e = 0; e < 8; e++) { a[e] = lo[e]; a[e + 8] = hi[e]; }
    qf[s] = a;
  }

  // Copy mapping: 2 x 16B chunks per thread per tile.
  // K tile: chunk c -> key = c>>3, dchunk = c&7.  V tile: d = c>>2, kc = c&3.
  int ck0 = tid * 2, ck1 = tid * 2 + 1;
  uint32_t kL0 = lds_addr(&ldsK[0][0][0]) + ck0 * 16;
  uint32_t kL1 = lds_addr(&ldsK[0][0][0]) + ck1 * 16;
  uint32_t vL0 = lds_addr(&ldsV[0][0][0]) + ck0 * 16;
  uint32_t vL1 = lds_addr(&ldsV[0][0][0]) + ck1 * 16;
  const __bf16* gK0 = kp + (size_t)(ck0 >> 3) * C3 + (ck0 & 7) * 8;
  const __bf16* gK1 = kp + (size_t)(ck1 >> 3) * C3 + (ck1 & 7) * 8;
  const __bf16* gV0 = vp + (size_t)(ck0 >> 2) * TT + (ck0 & 3) * 8;
  const __bf16* gV1 = vp + (size_t)(ck1 >> 2) * TT + (ck1 & 3) * 8;

  float m_run[8], l_run[8];
  v8f o[4] = {};
#pragma unroll
  for (int r = 0; r < 8; r++) { m_run[r] = -1e30f; l_run[r] = 0.0f; }

  int nkt = (qbase + 64) / 32;

  // Prologue: key-tile 0 -> buffer 0.
  async_copy16(kL0, gK0);
  async_copy16(kL1, gK1);
  async_copy16(vL0, gV0);
  async_copy16(vL1, gV1);

  for (int kt = 0; kt < nkt; kt++) {
    async_wait();
    __syncthreads();
    if (kt + 1 < nkt) {
      int nb = (kt + 1) & 1;
      int kt32 = (kt + 1) * 32;
      async_copy16(kL0 + nb * 4096, gK0 + (size_t)kt32 * C3);
      async_copy16(kL1 + nb * 4096, gK1 + (size_t)kt32 * C3);
      async_copy16(vL0 + nb * 4096, gV0 + kt32);
      async_copy16(vL1 + nb * 4096, gV1 + kt32);
    }
    int cb = kt & 1;

    // S = Q @ K^T : two 16-key tiles, K-dim = d (2 steps of 32)
    v8f s0 = {}, s1 = {};
    {
      v16bf k00 = load_fragB(&ldsK[cb][fr][kbB]);
      v16bf k01 = load_fragB(&ldsK[cb][fr][32 + kbB]);
      s0 = wmma_bf16(qf[0], k00, s0);
      s0 = wmma_bf16(qf[1], k01, s0);
      v16bf k10 = load_fragB(&ldsK[cb][16 + fr][kbB]);
      v16bf k11 = load_fragB(&ldsK[cb][16 + fr][32 + kbB]);
      s1 = wmma_bf16(qf[0], k10, s1);
      s1 = wmma_bf16(qf[1], k11, s1);
    }

    // Online softmax across this lane-half's 8 rows.
#pragma unroll
    for (int r = 0; r < 8; r++) {
      int rowg = qrow0 + r + half8;
      int c0 = kt * 32 + fr;
      int c1 = c0 + 16;
      float x0 = (c0 <= rowg) ? s0[r] : -1e30f;
      float x1 = (c1 <= rowg) ? s1[r] : -1e30f;
      float mnew = fmaxf(m_run[r], redmax16(fmaxf(x0, x1)));
      float p0 = __expf(x0 - mnew);
      float p1 = __expf(x1 - mnew);
      float sc = __expf(m_run[r] - mnew);
      l_run[r] = l_run[r] * sc + redsum16(p0 + p1);
      m_run[r] = mnew;
#pragma unroll
      for (int nt = 0; nt < 4; nt++) o[nt][r] *= sc;
      ldsP[wave][r + half8][fr]      = f2bf(p0);
      ldsP[wave][r + half8][fr + 16] = f2bf(p1);
    }
    __syncthreads();

    // O += P @ V : K-dim = 32 keys, 4 output d-tiles.
    v16bf pf = load_fragA(&ldsP[wave][fr][kbA]);
#pragma unroll
    for (int nt = 0; nt < 4; nt++) {
      v16bf vf = load_fragB(&ldsV[cb][nt * 16 + fr][kbB]);
      o[nt] = wmma_bf16(pf, vf, o[nt]);
    }
    __syncthreads();   // reads done before buffer reuse at kt+2
  }

  __bf16* yb = y + (size_t)(b * TT) * N_EMBD + h * HEAD_DIM;
#pragma unroll
  for (int r = 0; r < 8; r++) {
    float inv = 1.0f / l_run[r];
    int rowg = qrow0 + r + half8;
#pragma unroll
    for (int nt = 0; nt < 4; nt++) {
      yb[(size_t)rowg * N_EMBD + nt * 16 + fr] = f2bf(o[nt][r] * inv);
    }
  }
}

// ---------------------------------------------------------------------------
extern "C" void kernel_launch(void* const* d_in, const int* in_sizes, int n_in,
                              void* d_out, int out_size, void* d_ws,
                              size_t ws_size, hipStream_t stream) {
  const float* x       = (const float*)d_in[0];
  const float* ln1_g   = (const float*)d_in[1];
  const float* ln1_b   = (const float*)d_in[2];
  const float* W_attn  = (const float*)d_in[3];
  const float* b_attn  = (const float*)d_in[4];
  const float* W_cproj = (const float*)d_in[5];
  const float* b_cproj = (const float*)d_in[6];
  const float* ln2_g   = (const float*)d_in[7];
  const float* ln2_b   = (const float*)d_in[8];
  const float* W_fc    = (const float*)d_in[9];
  const float* b_fc    = (const float*)d_in[10];
  const float* W_mproj = (const float*)d_in[11];
  const float* b_mproj = (const float*)d_in[12];
  float* out = (float*)d_out;

  const size_t R = (size_t)ROWS;
  char* p = (char*)d_ws;
  float*  x2   = (float*)p;  p += R * N_EMBD * 4;        // residual stream (f32)
  __bf16* h1b  = (__bf16*)p; p += R * N_EMBD * 2;
  __bf16* qkvb = (__bf16*)p; p += R * C3 * 2;
  __bf16* vTb  = (__bf16*)p; p += R * N_EMBD * 2;        // [B,H,D,T]
  __bf16* yb   = (__bf16*)p; p += R * N_EMBD * 2;
  __bf16* h2b  = (__bf16*)p; p += R * N_EMBD * 2;
  __bf16* fcb  = (__bf16*)p; p += R * C4 * 2;
  __bf16* wtA  = (__bf16*)p; p += (size_t)C3 * N_EMBD * 2;
  __bf16* wtC  = (__bf16*)p; p += (size_t)N_EMBD * N_EMBD * 2;
  __bf16* wtF  = (__bf16*)p; p += (size_t)C4 * N_EMBD * 2;
  __bf16* wtM  = (__bf16*)p; p += (size_t)N_EMBD * C4 * 2;

  // One-time weight convert + transpose to bf16 [N][K].
  wconv_kernel<<<(N_EMBD * C3) / 256, 256, 0, stream>>>(W_attn, wtA, N_EMBD, C3);
  wconv_kernel<<<(N_EMBD * N_EMBD) / 256, 256, 0, stream>>>(W_cproj, wtC, N_EMBD, N_EMBD);
  wconv_kernel<<<(N_EMBD * C4) / 256, 256, 0, stream>>>(W_fc, wtF, N_EMBD, C4);
  wconv_kernel<<<(C4 * N_EMBD) / 256, 256, 0, stream>>>(W_mproj, wtM, C4, N_EMBD);

  // h1 = LN1(x)  (bf16)
  ln_kernel<<<ROWS, 256, 0, stream>>>(x, ln1_g, ln1_b, h1b);
  // qkv = h1 @ W_attn + b_attn  (Q pre-scaled, V also written [b,h,d,t])
  gemm_kernel<2><<<dim3(C3 / 64, ROWS / 128), 512, 0, stream>>>(
      h1b, wtA, b_attn, nullptr, nullptr, qkvb, vTb, ROWS, C3, N_EMBD);
  // y = causal softmax(q k^T) v  (bf16)
  attn_kernel<<<dim3(TT / 64, N_HEAD, BB), 128, 0, stream>>>(qkvb, vTb, yb);
  // x2 = x + y @ W_cproj + b_cproj  (f32)
  gemm_kernel<0><<<dim3(N_EMBD / 64, ROWS / 128), 512, 0, stream>>>(
      yb, wtC, b_cproj, x, x2, nullptr, nullptr, ROWS, N_EMBD, N_EMBD);
  // h2 = LN2(x2)  (bf16)
  ln_kernel<<<ROWS, 256, 0, stream>>>(x2, ln2_g, ln2_b, h2b);
  // fc = gelu(h2 @ W_fc + b_fc)  (bf16)
  gemm_kernel<1><<<dim3(C4 / 64, ROWS / 128), 512, 0, stream>>>(
      h2b, wtF, b_fc, nullptr, nullptr, fcb, nullptr, ROWS, C4, N_EMBD);
  // out = x2 + fc @ W_mproj + b_mproj  (f32)
  gemm_kernel<0><<<dim3(N_EMBD / 64, ROWS / 128), 512, 0, stream>>>(
      fcb, wtM, b_mproj, x2, out, nullptr, nullptr, ROWS, N_EMBD, C4);
}
